// MultiHeadAttention_39187281609297
// MI455X (gfx1250) — matrix operand order repaired and checked by reading, not compiled, not run on hardware
//
#include <hip/hip_runtime.h>

typedef __attribute__((ext_vector_type(16))) _Float16 v16h;
typedef __attribute__((ext_vector_type(8)))  float    v8f;

constexpr int BB = 2, SS = 4096, CC = 512, HH = 8, DD = 64;
constexpr int LDP = SS + 4;   // padded LDS row stride (4100 mod 64 = 4 -> no bank conflicts)

// ---------------------------------------------------------------------------
// WMMA fragment loaders (CDNA5 wave32, V_WMMA_F32_16X16X32_F16; A/B use a contiguous k
// order, not the ISA table order, the same in A and B):
//   A 16x32 (MxK): lanes 0-15 -> rows M=0..15 with K=0..15 in v[0..15];
//                  lanes 16-31 -> same rows, K=16..31.
//   B 32x16 (KxN): N = lane&15; lanes 0-15 hold K=0..15 (v[0..15]),
//                  lanes 16-31 hold K=16..31.
//   C/D 16x16:     VGPR r -> M = r + 8*(lane>=16), N = lane&15.
// ---------------------------------------------------------------------------

template <typename T>
__device__ __forceinline__ v16h load_frag_A(const T* __restrict__ X, int ld, int k0) {
  const int lane = threadIdx.x & 31;
  const int row  = lane & 15;
  const int seg  = (lane >> 4) * 16;  // 0 or 16
  const T* p = X + (size_t)row * ld + k0;
  v16h a;
#pragma unroll
  for (int i = 0; i < 16; ++i) a[i] = (_Float16)p[seg + i];
  return a;
}

// B = W^T where W is stored row-major [N x K]; element (k, n) = W[n, k].
template <typename T>
__device__ __forceinline__ v16h load_frag_B(const T* __restrict__ W, int ld, int k0) {
  const int lane = threadIdx.x & 31;
  const int n    = lane & 15;
  const int kseg = (lane >> 4) * 16;  // 0 or 16
  const T* p = W + (size_t)n * ld + k0 + kseg;
  v16h b;
#pragma unroll
  for (int i = 0; i < 16; ++i) b[i] = (_Float16)p[i];
  return b;
}

// ---------------------------------------------------------------------------
// Y = X @ W^T + bias.  X: [M,K] (f32 or f16), W: [N,K] f32.
// SPLIT_HEADS=1: write f16 into split-head layout [B,H,S,D], scaled.
// SPLIT_HEADS=0: write f32 row-major [M,N].
// One wave per 16x16 tile; 4 waves per block share an M-tile (A hits L0/L2).
// ---------------------------------------------------------------------------
template <typename TIN, int SPLIT_HEADS>
__global__ void __launch_bounds__(128)
gemm_xwt_wmma(const TIN* __restrict__ X, const float* __restrict__ W,
              const float* __restrict__ bias, float* __restrict__ Yf,
              _Float16* __restrict__ Yh, float scale, int M, int N, int K) {
  const int wave  = threadIdx.x >> 5;
  const int ntile = blockIdx.x * 4 + wave;
  const int mtile = blockIdx.y;
  const TIN*  Xt = X + (size_t)mtile * 16 * K;
  const float* Wt = W + (size_t)ntile * 16 * K;

  v8f acc = {};
  for (int k0 = 0; k0 < K; k0 += 32) {
    v16h a = load_frag_A(Xt, K, k0);
    v16h b = load_frag_B(Wt, K, k0);
    acc = __builtin_amdgcn_wmma_f32_16x16x32_f16(false, a, false, b,
                                                 (short)0, acc, false, false);
  }

  const int lane = threadIdx.x & 31;
  const int col  = lane & 15;
  const int rb   = (lane >> 4) * 8;
#pragma unroll
  for (int r = 0; r < 8; ++r) {
    const int grow = mtile * 16 + rb + r;
    const int gcol = ntile * 16 + col;
    const float v = acc[r] + bias[gcol];
    if (SPLIT_HEADS) {
      const int bIdx = grow >> 12;       // / S (4096)
      const int s    = grow & (SS - 1);
      const int head = gcol >> 6;        // / D (64)
      const int d    = gcol & (DD - 1);
      Yh[((((size_t)bIdx * HH + head) << 12) + s) * DD + d] = (_Float16)(v * scale);
    } else {
      Yf[(size_t)grow * N + gcol] = v;
    }
  }
}

// ---------------------------------------------------------------------------
// Attention: one block (4 waves) per (b, h, 16-query tile).
// Full 16 x 4096 f32 score strip lives in LDS (263 KB of the 320 KB WGP LDS,
// row stride padded to 4100 words so no phase hits LDS bank conflicts).
// Phase 1: raw scores = (Q/sqrt(D)) K^T via WMMA, waves stripe key tiles,
//          next K-tile prefetched (global_prefetch_b8).
// Phase 2: flat coalesced mask add (mask rows for this block are contiguous),
//          per-row max / exp / sum in LDS, then flat coalesced normalize with
//          the 1.07 GB attn write fully streamed (512 B per 128 lanes).
// Phase 3: out = P V via WMMA (A converted f32->f16 out of LDS), f16 concat.
// ---------------------------------------------------------------------------
__global__ void __launch_bounds__(128)
attn_wmma(const _Float16* __restrict__ qh, const _Float16* __restrict__ kh,
          const _Float16* __restrict__ vh, const float* __restrict__ mask,
          float* __restrict__ attn_out, _Float16* __restrict__ concat) {
  extern __shared__ float lds[];
  float* scores = lds;               // 16 * LDP
  float* red    = lds + 16 * LDP;    // 128
  float* rowm   = red + 128;         // 16
  float* rowinv = rowm + 16;         // 16

  const int qt   = blockIdx.x;
  const int h    = blockIdx.y;
  const int b    = blockIdx.z;
  const int wave = threadIdx.x >> 5;
  const int lane = threadIdx.x & 31;
  const int col  = lane & 15;
  const int rb   = (lane >> 4) * 8;

  const size_t headOff = ((size_t)b * HH + h) * SS * DD;
  const _Float16* Q  = qh + headOff + (size_t)qt * 16 * DD;
  const _Float16* Km = kh + headOff;
  const _Float16* V  = vh + headOff;

  // Loop-invariant Q fragments (16 x 64 = two K=32 chunks), q pre-scaled.
  const v16h qa0 = load_frag_A(Q, DD, 0);
  const v16h qa1 = load_frag_A(Q, DD, 32);

  // ---- Phase 1: raw scores into LDS ----
  for (int kt = wave; kt < SS / 16; kt += 4) {
    const _Float16* Kt = Km + (size_t)kt * 16 * DD;
    if (kt + 4 < SS / 16)   // hide latency: prefetch this wave's next K tile
      __builtin_prefetch(Km + (size_t)(kt + 4) * 16 * DD + (lane & 15) * DD, 0, 0);
    v16h kb0 = load_frag_B(Kt, DD, 0);
    v16h kb1 = load_frag_B(Kt, DD, 32);
    v8f c = {};
    c = __builtin_amdgcn_wmma_f32_16x16x32_f16(false, qa0, false, kb0, (short)0, c, false, false);
    c = __builtin_amdgcn_wmma_f32_16x16x32_f16(false, qa1, false, kb1, (short)0, c, false, false);
#pragma unroll
    for (int r = 0; r < 8; ++r)
      scores[(rb + r) * LDP + kt * 16 + col] = c[r];
  }
  __syncthreads();

  // ---- Phase 2a: additive mask, flat & fully coalesced ----
  const float* maskBlk = mask + (size_t)(qt * 16) * SS;   // contiguous [16][S]
  for (int idx = threadIdx.x; idx < 16 * SS; idx += 128) {
    const int r = idx >> 12;          // / S
    const int c2 = idx & (SS - 1);
    scores[r * LDP + c2] += maskBlk[idx] * -1.0e12f;
  }
  __syncthreads();

  // ---- Phase 2b: softmax stats (8 threads per row, LDS only) ----
  const int row = threadIdx.x >> 3;
  const int sub = threadIdx.x & 7;
  float m = -3.0e38f;
  for (int k2 = sub; k2 < SS; k2 += 8) m = fmaxf(m, scores[row * LDP + k2]);
  red[threadIdx.x] = m;
  __syncthreads();
  if (sub == 0) {
    float mm = red[row * 8];
#pragma unroll
    for (int i = 1; i < 8; ++i) mm = fmaxf(mm, red[row * 8 + i]);
    rowm[row] = mm;
  }
  __syncthreads();
  const float mm = rowm[row];
  float sacc = 0.0f;
  for (int k2 = sub; k2 < SS; k2 += 8) {
    const float e = __expf(scores[row * LDP + k2] - mm);
    scores[row * LDP + k2] = e;
    sacc += e;
  }
  red[threadIdx.x] = sacc;
  __syncthreads();
  if (sub == 0) {
    float ssum = 0.0f;
#pragma unroll
    for (int i = 0; i < 8; ++i) ssum += red[row * 8 + i];
    rowinv[row] = 1.0f / ssum;
  }
  __syncthreads();

  // ---- Phase 2c: normalize + stream attn to HBM, flat & fully coalesced ----
  float* aout = attn_out + (((size_t)b * HH + h) * SS + (size_t)qt * 16) * SS;
  for (int idx = threadIdx.x; idx < 16 * SS; idx += 128) {
    const int r = idx >> 12;
    const int c2 = idx & (SS - 1);
    const float p = scores[r * LDP + c2] * rowinv[r];
    scores[r * LDP + c2] = p;
    aout[idx] = p;                    // the mandated 1.07 GB output write
  }
  __syncthreads();

  // ---- Phase 3: out = P V; each wave owns 16 of the 64 head dims ----
  v8f o = {};
  const int dbase = wave * 16;
  for (int kc = 0; kc < SS; kc += 32) {
    v16h pa = load_frag_A(scores, LDP, kc);   // f32 LDS -> f16 frag
    v16h vb;
    const int kseg = (lane >> 4) * 16;
#pragma unroll
    for (int i = 0; i < 16; ++i)
      vb[i] = V[(size_t)(kc + kseg + i) * DD + dbase + col];
    o = __builtin_amdgcn_wmma_f32_16x16x32_f16(false, pa, false, vb, (short)0, o, false, false);
  }
#pragma unroll
  for (int r = 0; r < 8; ++r) {
    const int srow = qt * 16 + rb + r;
    concat[((size_t)b * SS + srow) * CC + h * DD + dbase + col] = (_Float16)o[r];
  }
}

// ---------------------------------------------------------------------------
extern "C" void kernel_launch(void* const* d_in, const int* in_sizes, int n_in,
                              void* d_out, int out_size, void* d_ws, size_t ws_size,
                              hipStream_t stream) {
  (void)in_sizes; (void)n_in; (void)out_size; (void)ws_size;
  const float* q    = (const float*)d_in[0];
  const float* k    = (const float*)d_in[1];
  const float* v    = (const float*)d_in[2];
  const float* mask = (const float*)d_in[3];
  const float* wq   = (const float*)d_in[4];
  const float* bq   = (const float*)d_in[5];
  const float* wk   = (const float*)d_in[6];
  const float* bk   = (const float*)d_in[7];
  const float* wv   = (const float*)d_in[8];
  const float* bv   = (const float*)d_in[9];
  const float* wo   = (const float*)d_in[10];
  const float* bo   = (const float*)d_in[11];

  float* out  = (float*)d_out;                    // [B,S,C]
  float* attn = out + (size_t)BB * SS * CC;       // [B,H,S,S]

  const size_t headElems = (size_t)BB * HH * SS * DD;  // 4.19M halves each
  _Float16* qh     = (_Float16*)d_ws;
  _Float16* kh     = qh + headElems;
  _Float16* vh     = kh + headElems;
  _Float16* concat = vh + headElems;              // [B,S,C] f16; 32 MB total ws

  const int M = BB * SS;                          // 8192
  const dim3 blk(128);
  const dim3 gproj(CC / 64, M / 16);              // (8, 512)

  const float qscale = 0.125f;                    // 1/sqrt(D)
  gemm_xwt_wmma<float, 1><<<gproj, blk, 0, stream>>>(q, wq, bq, nullptr, qh, qscale, M, CC, CC);
  gemm_xwt_wmma<float, 1><<<gproj, blk, 0, stream>>>(k, wk, bk, nullptr, kh, 1.0f,   M, CC, CC);
  gemm_xwt_wmma<float, 1><<<gproj, blk, 0, stream>>>(v, wv, bv, nullptr, vh, 1.0f,   M, CC, CC);

  const size_t ldsBytes = (size_t)(16 * LDP + 128 + 32) * sizeof(float);  // ~263 KB
  hipError_t attrErr = hipFuncSetAttribute(
      (const void*)attn_wmma, hipFuncAttributeMaxDynamicSharedMemorySize, (int)ldsBytes);
  (void)attrErr;
  const dim3 gattn(SS / 16, HH, BB);              // (256, 8, 2)
  attn_wmma<<<gattn, blk, ldsBytes, stream>>>(qh, kh, vh, mask, attn, concat);

  gemm_xwt_wmma<_Float16, 0><<<gproj, blk, 0, stream>>>(concat, wo, bo, out, nullptr, 1.0f, M, CC, CC);
}
